// GINBert_32160715112911
// MI455X (gfx1250) — matrix operand order repaired and checked by reading, not compiled, not run on hardware
//
#include <hip/hip_runtime.h>
#include <hip/hip_bf16.h>
#include <cstdint>

// ---------------- problem constants ----------------
#define B_      128
#define NPG_    64
#define N_      (B_ * NPG_)      // 8192 nodes
#define E_      (4 * N_)         // 32768 edges
#define MAXN_   69
#define L_      512
#define DPROT_  1280
#define DIM_    32
#define NFX_    78
#define SPAD_   80               // 69 padded to 80 (multiple of 16)
#define LDA_    48               // halfs per LDS A row (32 data + pad, 96B: 32B-aligned frags)
#define LDB_    48               // halfs per LDS Bt row

typedef __attribute__((ext_vector_type(16))) _Float16 v16h;
typedef __attribute__((ext_vector_type(4)))  _Float16 v4h;
typedef __attribute__((ext_vector_type(2)))  _Float16 v2h;
typedef __attribute__((ext_vector_type(8)))  float    v8f;
typedef __attribute__((ext_vector_type(4)))  float    v4f;

// Swap bits 3<->4 of a K index: makes each lane's 16-half WMMA A-fragment
// (K {0..7,16..23} / {8..15,24..31}) a contiguous 32B run in LDS.
__device__ __forceinline__ int swap34(int c) {
  return (c & ~24) | (((c >> 4) & 1) << 3) | (((c >> 3) & 1) << 4);
}

__device__ __forceinline__ float lrelu(float x) { return x > 0.f ? x : 0.01f * x; }

// ---------------- utility ----------------
__global__ void fill0_kernel(float* p, int n) {
  int i = blockIdx.x * 256 + threadIdx.x;
  if (i < n) p[i] = 0.f;
}

// ---------------- GIN ----------------
__global__ void gin_agg_kernel(const float* __restrict__ h, const int* __restrict__ src,
                               const int* __restrict__ dst, float* __restrict__ agg, int dim) {
  int e = blockIdx.x, d = threadIdx.x;
  if (d < dim) atomicAdd(&agg[(size_t)dst[e] * dim + d], h[(size_t)src[e] * dim + d]);
}

__global__ __launch_bounds__(256) void gin_mlp_kernel(
    const float* __restrict__ h, const float* __restrict__ agg,
    const float* __restrict__ w1, const float* __restrict__ b1,
    const float* __restrict__ w2, const float* __restrict__ b2,
    float* __restrict__ z, int fin) {
  int n = blockIdx.x * blockDim.x + threadIdx.x;
  if (n >= N_) return;
  float zin[NFX_];
  for (int i = 0; i < fin; ++i) zin[i] = h[(size_t)n * fin + i] + agg[(size_t)n * fin + i];
  float h1[DIM_];
  for (int j = 0; j < DIM_; ++j) {
    float a = b1[j];
    for (int i = 0; i < fin; ++i) a += zin[i] * w1[i * DIM_ + j];
    h1[j] = lrelu(a);
  }
  for (int j = 0; j < DIM_; ++j) {
    float a = b2[j];
    for (int i = 0; i < DIM_; ++i) a += h1[i] * w2[i * DIM_ + j];
    z[(size_t)n * DIM_ + j] = lrelu(a);
  }
}

__global__ __launch_bounds__(256) void bn_stats_kernel(const float* __restrict__ z, float* __restrict__ stats) {
  int c = blockIdx.x;
  float s = 0.f, s2 = 0.f;
  for (int i = threadIdx.x; i < N_; i += 256) {
    float v = z[(size_t)i * DIM_ + c];
    s += v; s2 += v * v;
  }
  __shared__ float sh[256], sh2[256];
  sh[threadIdx.x] = s; sh2[threadIdx.x] = s2;
  __syncthreads();
  for (int o = 128; o > 0; o >>= 1) {
    if (threadIdx.x < o) { sh[threadIdx.x] += sh[threadIdx.x + o]; sh2[threadIdx.x] += sh2[threadIdx.x + o]; }
    __syncthreads();
  }
  if (threadIdx.x == 0) {
    float mu = sh[0] / (float)N_;
    stats[c] = mu;
    stats[DIM_ + c] = sh2[0] / (float)N_ - mu * mu;
  }
}

__global__ void bn_apply_kernel(const float* __restrict__ z, const float* __restrict__ stats,
                                const float* __restrict__ g, const float* __restrict__ bt,
                                float* __restrict__ out) {
  int i = blockIdx.x * 256 + threadIdx.x;
  if (i >= N_ * DIM_) return;
  int c = i & (DIM_ - 1);
  out[i] = g[c] * (z[i] - stats[c]) * rsqrtf(stats[DIM_ + c] + 1e-5f) + bt[c];
}

// ---------------- xd = lrelu(dense @ fc1 + b) : [B,69,128] ----------------
__global__ __launch_bounds__(128) void xd_kernel(const float* __restrict__ h,
                                                 const float* __restrict__ w, const float* __restrict__ b,
                                                 float* __restrict__ xd) {
  int bs = blockIdx.x;               // b*69 + s
  int gb = bs / MAXN_, s = bs % MAXN_;
  int j = threadIdx.x;
  float a = b[j];
  if (s < NPG_) {
    const float* hr = h + (size_t)(gb * NPG_ + s) * DIM_;
    for (int i = 0; i < DIM_; ++i) a += hr[i] * w[i * 128 + j];
  }
  xd[(size_t)bs * 128 + j] = lrelu(a);
}

// ---------------- protein GEMM: traw[b,l,n] = P[tid[b]] @ ab_w  (WMMA f16) ----
// Double-buffered LDS (1 barrier / K-step); all 8 B fragments preloaded to
// registers so DS loads clause up and waits are partial.
__global__ __launch_bounds__(256) void prot_gemm_kernel(
    const float* __restrict__ prot, const int* __restrict__ tid,
    const float* __restrict__ abw, float* __restrict__ traw) {
  __shared__ __align__(32) _Float16 As[2][128 * LDA_];  // swizzled: As[m][swap34(k)]
  __shared__ __align__(32) _Float16 Bt[2][128 * LDB_];  // transposed: Bt[n][k]
  int b = blockIdx.x, mt = blockIdx.y;
  int t = threadIdx.x, lane = t & 31, wv = t >> 5;
  const float* A = prot + (size_t)tid[b] * L_ * DPROT_ + (size_t)mt * 128 * DPROT_;
  int mrow = (wv << 4) + (lane & 15);
  int hofs = (lane >> 4) << 4;      // 0 or 16 halfs into the fragment run
  v8f acc[8];
#pragma unroll
  for (int i = 0; i < 8; ++i)
#pragma unroll
    for (int j = 0; j < 8; ++j) acc[i][j] = 0.f;

  for (int ki = 0; ki < DPROT_ / 32; ++ki) {
    int k0 = ki << 5, p = ki & 1;
    const float* Apre = A + ((k0 + 32 < DPROT_) ? (k0 + 32) : k0);
    // stage A: 128x32 floats as 1024 float4 chunks -> swizzled f16
    for (int ch = t; ch < 1024; ch += 256) {
      int r = ch >> 3, c4 = (ch & 7) << 2;
      v4f f = *(const v4f*)(A + (size_t)r * DPROT_ + k0 + c4);
      v4h hv;
      hv[0] = (_Float16)f[0]; hv[1] = (_Float16)f[1];
      hv[2] = (_Float16)f[2]; hv[3] = (_Float16)f[3];
      *(v4h*)(&As[p][r * LDA_ + swap34(c4)]) = hv;
      __builtin_prefetch(Apre + (size_t)r * DPROT_ + c4, 0, 1);
    }
    // stage B transposed: Bt[n][k] from abw[k0+k][n]
    for (int i = t; i < 2048; i += 256) {
      int c = i & 127, r = (i >> 7) << 1;
      v2h hv;
      hv[0] = (_Float16)abw[(size_t)(k0 + r) * 128 + c];
      hv[1] = (_Float16)abw[(size_t)(k0 + r + 1) * 128 + c];
      *(v2h*)(&Bt[p][c * LDB_ + swap34(r)]) = hv;
    }
    __syncthreads();
    v16h af = *(const v16h*)(&As[p][mrow * LDA_ + hofs]);
    v16h bf[8];
#pragma unroll
    for (int nt = 0; nt < 8; ++nt)
      bf[nt] = *(const v16h*)(&Bt[p][((nt << 4) + (lane & 15)) * LDB_ + hofs]);
#pragma unroll
    for (int nt = 0; nt < 8; ++nt)
      acc[nt] = __builtin_amdgcn_wmma_f32_16x16x32_f16(false, af, false, bf[nt],
                                                       (short)0, acc[nt], false, false);
    // no trailing barrier: next iteration writes the other LDS buffer, whose
    // last readers were ordered by the *previous* barrier.
  }
  float* out = traw + ((size_t)b * L_ + mt * 128 + wv * 16) * 128;
  int rofs = (lane >> 4) * 8, col0 = lane & 15;
#pragma unroll
  for (int nt = 0; nt < 8; ++nt)
#pragma unroll
    for (int r = 0; r < 8; ++r)
      out[(size_t)(r + rofs) * 128 + nt * 16 + col0] = acc[nt][r];
}

// ---------------- bias + layernorm + lrelu over 128-wide rows ----------------
__global__ __launch_bounds__(256) void ln_kernel(const float* __restrict__ traw,
                                                 const float* __restrict__ bias,
                                                 const float* __restrict__ g,
                                                 const float* __restrict__ bt,
                                                 float* __restrict__ out) {
  int row = blockIdx.x * 8 + (threadIdx.x >> 5);
  int lane = threadIdx.x & 31;
  const float* r = traw + (size_t)row * 128;
  float v[4], s = 0.f, s2 = 0.f;
#pragma unroll
  for (int i = 0; i < 4; ++i) {
    int c = lane + 32 * i;
    v[i] = r[c] + bias[c];
    s += v[i]; s2 += v[i] * v[i];
  }
  for (int o = 16; o > 0; o >>= 1) { s += __shfl_xor(s, o, 32); s2 += __shfl_xor(s2, o, 32); }
  float mu = s * (1.f / 128.f);
  float rs = rsqrtf(s2 * (1.f / 128.f) - mu * mu + 1e-5f);
  float* op = out + (size_t)row * 128;
#pragma unroll
  for (int i = 0; i < 4; ++i) {
    int c = lane + 32 * i;
    op[c] = lrelu(g[c] * (v[i] - mu) * rs + bt[c]);
  }
}

// ---------------- WbxT[b,s,k] = sum_m W_b[k,m] * xd[b,s,m]  (s padded to 80) --
__global__ __launch_bounds__(128) void wbx_kernel(const float* __restrict__ xd,
                                                  const float* __restrict__ Wb,
                                                  float* __restrict__ wbxT) {
  int bs = blockIdx.x;               // b*80 + s
  int gb = bs / SPAD_, s = bs % SPAD_;
  int k = threadIdx.x;
  float a = 0.f;
  if (s < MAXN_) {
    const float* xr = xd + ((size_t)gb * MAXN_ + s) * 128;
    for (int m = 0; m < 128; ++m) a += Wb[k * 128 + m] * xr[m];
  }
  wbxT[(size_t)bs * 128 + k] = a;
}

// ---------------- C = tanh(t @ WbxT^T) : [B,512,80]  (WMMA f16) --------------
__global__ __launch_bounds__(256) void cgemm_kernel(const float* __restrict__ tmat,
                                                    const float* __restrict__ wbxT,
                                                    float* __restrict__ Cb) {
  __shared__ __align__(32) _Float16 As[2][128 * LDA_];   // swizzled 128 x 32
  __shared__ __align__(32) _Float16 Bt[2][SPAD_ * LDB_]; // Bt[n][k] (K-major in wbxT)
  int b = blockIdx.x, mt = blockIdx.y;
  int t = threadIdx.x, lane = t & 31, wv = t >> 5;
  const float* A = tmat + ((size_t)b * L_ + mt * 128) * 128;
  const float* Bg = wbxT + (size_t)b * SPAD_ * 128;
  int mrow = (wv << 4) + (lane & 15);
  int hofs = (lane >> 4) << 4;
  v8f acc[5];
#pragma unroll
  for (int i = 0; i < 5; ++i)
#pragma unroll
    for (int j = 0; j < 8; ++j) acc[i][j] = 0.f;

  for (int ki = 0; ki < 4; ++ki) {
    int k0 = ki << 5, p = ki & 1;
    for (int ch = t; ch < 1024; ch += 256) {
      int r = ch >> 3, c4 = (ch & 7) << 2;
      v4f f = *(const v4f*)(A + (size_t)r * 128 + k0 + c4);
      v4h hv;
      hv[0] = (_Float16)f[0]; hv[1] = (_Float16)f[1];
      hv[2] = (_Float16)f[2]; hv[3] = (_Float16)f[3];
      *(v4h*)(&As[p][r * LDA_ + swap34(c4)]) = hv;
    }
    for (int ch = t; ch < SPAD_ * 8; ch += 256) {   // 80 rows x 8 float4 chunks
      int n = ch >> 3, c4 = (ch & 7) << 2;
      v4f f = *(const v4f*)(Bg + (size_t)n * 128 + k0 + c4);
      v4h hv;
      hv[0] = (_Float16)f[0]; hv[1] = (_Float16)f[1];
      hv[2] = (_Float16)f[2]; hv[3] = (_Float16)f[3];
      *(v4h*)(&Bt[p][n * LDB_ + swap34(c4)]) = hv;
    }
    __syncthreads();
    v16h af = *(const v16h*)(&As[p][mrow * LDA_ + hofs]);
    v16h bf[5];
#pragma unroll
    for (int nt = 0; nt < 5; ++nt)
      bf[nt] = *(const v16h*)(&Bt[p][((nt << 4) + (lane & 15)) * LDB_ + hofs]);
#pragma unroll
    for (int nt = 0; nt < 5; ++nt)
      acc[nt] = __builtin_amdgcn_wmma_f32_16x16x32_f16(false, af, false, bf[nt],
                                                       (short)0, acc[nt], false, false);
  }
  float* out = Cb + ((size_t)b * L_ + mt * 128 + wv * 16) * SPAD_;
  int rofs = (lane >> 4) * 8, col0 = lane & 15;
#pragma unroll
  for (int nt = 0; nt < 5; ++nt)
#pragma unroll
    for (int r = 0; r < 8; ++r)
      out[(size_t)(r + rofs) * SPAD_ + nt * 16 + col0] = tanhf(acc[nt][r]);
}

// ---------------- Wc_x[b,d,s], Wp_t[b,d,l] ----------------
__global__ void wcx_kernel(const float* __restrict__ xd, const float* __restrict__ Wc,
                           float* __restrict__ wcx) {
  int idx = blockIdx.x * 256 + threadIdx.x;
  if (idx >= B_ * 32 * MAXN_) return;
  int s = idx % MAXN_, d = (idx / MAXN_) % 32, gb = idx / (MAXN_ * 32);
  const float* xr = xd + ((size_t)gb * MAXN_ + s) * 128;
  const float* w = Wc + d * 128;
  float a = 0.f;
  for (int k = 0; k < 128; ++k) a += w[k] * xr[k];
  wcx[idx] = a;
}

__global__ void wpt_kernel(const float* __restrict__ tmat, const float* __restrict__ Wp,
                           float* __restrict__ wpt) {
  int idx = blockIdx.x * 256 + threadIdx.x;
  if (idx >= B_ * 32 * L_) return;
  int l = idx % L_, d = (idx / L_) % 32, gb = idx / (L_ * 32);
  const float* tr = tmat + ((size_t)gb * L_ + l) * 128;
  const float* w = Wp + d * 128;
  float a = 0.f;
  for (int k = 0; k < 128; ++k) a += w[k] * tr[k];
  wpt[idx] = a;
}

// ---------------- attention over compound (a_c, c) ----------------
__global__ __launch_bounds__(256) void attn_c_kernel(
    const float* __restrict__ xd, const float* __restrict__ wcxg,
    const float* __restrict__ wptg, const float* __restrict__ Cb,
    const float* __restrict__ whc, float* __restrict__ a_out, float* __restrict__ cvec) {
  int b = blockIdx.x, t = threadIdx.x;
  __shared__ float score[MAXN_];
  __shared__ float red;
  const float* wcx = wcxg + (size_t)b * 32 * MAXN_;
  const float* wpt = wptg + (size_t)b * 32 * L_;
  const float* C   = Cb + (size_t)b * L_ * SPAD_;
  if (t < MAXN_) score[t] = 0.f;
  __syncthreads();
  for (int idx = t; idx < 32 * MAXN_; idx += 256) {
    int d = idx / MAXN_, s = idx % MAXN_;
    float a = wcx[idx];
    const float* wl = wpt + (size_t)d * L_;
    for (int l = 0; l < L_; ++l) a += wl[l] * C[(size_t)l * SPAD_ + s];
    atomicAdd(&score[s], whc[d] * tanhf(a));
  }
  __syncthreads();
  if (t == 0) {
    float m = -1e30f;
    for (int s = 0; s < MAXN_; ++s) m = fmaxf(m, score[s]);
    float sum = 0.f;
    for (int s = 0; s < MAXN_; ++s) { float e = __expf(score[s] - m); score[s] = e; sum += e; }
    red = sum;
  }
  __syncthreads();
  float inv = 1.f / red;
  if (t < MAXN_) { score[t] *= inv; a_out[(size_t)b * MAXN_ + t] = score[t]; }
  __syncthreads();
  if (t < 128) {
    const float* xr = xd + (size_t)b * MAXN_ * 128;
    float a = 0.f;
    for (int s = 0; s < MAXN_; ++s) a += score[s] * xr[(size_t)s * 128 + t];
    cvec[(size_t)b * 128 + t] = a;
  }
}

// ---------------- attention over protein (a_p, pvec) ----------------
__global__ __launch_bounds__(256) void attn_p_kernel(
    const float* __restrict__ tmat, const float* __restrict__ wcxg,
    const float* __restrict__ wptg, const float* __restrict__ Cb,
    const float* __restrict__ whp, float* __restrict__ a_out, float* __restrict__ pv) {
  int b = blockIdx.x, t = threadIdx.x;
  __shared__ float score[L_];
  __shared__ float red[256];
  const float* wcx = wcxg + (size_t)b * 32 * MAXN_;
  const float* wpt = wptg + (size_t)b * 32 * L_;
  const float* C   = Cb + (size_t)b * L_ * SPAD_;
  for (int l = t; l < L_; l += 256) score[l] = 0.f;
  __syncthreads();
  for (int idx = t; idx < 32 * L_; idx += 256) {
    int d = idx / L_, l = idx % L_;
    float a = wpt[idx];
    const float* Cl = C + (size_t)l * SPAD_;
    const float* wd = wcx + (size_t)d * MAXN_;
    for (int s = 0; s < MAXN_; ++s) a += wd[s] * Cl[s];
    atomicAdd(&score[l], whp[d] * tanhf(a));
  }
  __syncthreads();
  float m = fmaxf(score[t], score[t + 256]);
  red[t] = m; __syncthreads();
  for (int o = 128; o > 0; o >>= 1) { if (t < o) red[t] = fmaxf(red[t], red[t + o]); __syncthreads(); }
  m = red[0]; __syncthreads();
  float e0 = __expf(score[t] - m), e1 = __expf(score[t + 256] - m);
  score[t] = e0; score[t + 256] = e1;
  red[t] = e0 + e1; __syncthreads();
  for (int o = 128; o > 0; o >>= 1) { if (t < o) red[t] += red[t + o]; __syncthreads(); }
  float inv = 1.f / red[0];
  __syncthreads();
  for (int l = t; l < L_; l += 256) { float v = score[l] * inv; score[l] = v; a_out[(size_t)b * L_ + l] = v; }
  __syncthreads();
  if (t < 128) {
    const float* tr = tmat + (size_t)b * L_ * 128;
    float a = 0.f;
    for (int l = 0; l < L_; ++l) a += score[l] * tr[(size_t)l * 128 + t];
    pv[(size_t)b * 128 + t] = a;
  }
}

// ---------------- MLP head ----------------
__global__ __launch_bounds__(256) void head_kernel(
    const float* __restrict__ cvec, const float* __restrict__ pv,
    const float* __restrict__ faw, const float* __restrict__ fab,
    const float* __restrict__ fbw, const float* __restrict__ fbb,
    const float* __restrict__ ow, const float* __restrict__ ob, float* __restrict__ out) {
  int b = blockIdx.x, t = threadIdx.x;
  __shared__ float cp[256];
  __shared__ float h1[1024];
  __shared__ float h2[256];
  cp[t] = (t < 128) ? cvec[(size_t)b * 128 + t] : pv[(size_t)b * 128 + (t - 128)];
  __syncthreads();
  for (int j = t; j < 1024; j += 256) {
    float a = fab[j];
    for (int i = 0; i < 256; ++i) a += cp[i] * faw[(size_t)i * 1024 + j];
    h1[j] = lrelu(a);
  }
  __syncthreads();
  {
    float a = fbb[t];
    for (int i = 0; i < 1024; ++i) a += h1[i] * fbw[(size_t)i * 256 + t];
    h2[t] = lrelu(a);
  }
  __syncthreads();
  if (t == 0) {
    float a = ob[0];
    for (int i = 0; i < 256; ++i) a += h2[i] * ow[i];
    out[b] = a;
  }
}

// ---------------- launcher ----------------
extern "C" void kernel_launch(void* const* d_in, const int* in_sizes, int n_in,
                              void* d_out, int out_size, void* d_ws, size_t ws_size,
                              hipStream_t stream) {
  (void)in_sizes; (void)n_in; (void)out_size; (void)ws_size;
  const float* x    = (const float*)d_in[0];
  const int*   ei   = (const int*)d_in[1];
  const int*   tid  = (const int*)d_in[2];
  const float* prot = (const float*)d_in[4];
  const float* fc1w = (const float*)d_in[35];
  const float* fc1b = (const float*)d_in[36];
  const float* abw  = (const float*)d_in[37];
  const float* abb  = (const float*)d_in[38];
  const float* lng  = (const float*)d_in[39];
  const float* lnb  = (const float*)d_in[40];
  const float* Wb   = (const float*)d_in[41];
  const float* Wc   = (const float*)d_in[42];
  const float* Wp   = (const float*)d_in[43];
  const float* whc  = (const float*)d_in[44];
  const float* whp  = (const float*)d_in[45];
  const float* faw  = (const float*)d_in[46];
  const float* fab  = (const float*)d_in[47];
  const float* fbw  = (const float*)d_in[48];
  const float* fbb  = (const float*)d_in[49];
  const float* ow   = (const float*)d_in[50];
  const float* ob   = (const float*)d_in[51];

  char* ws = (char*)d_ws;
  size_t off = 0;
  auto carve = [&](size_t bytes) -> float* {
    float* p = (float*)(ws + off);
    off += (bytes + 255) & ~(size_t)255;
    return p;
  };
  float* agg   = carve((size_t)N_ * NFX_ * 4);
  float* zbuf  = carve((size_t)N_ * DIM_ * 4);
  float* hA    = carve((size_t)N_ * DIM_ * 4);
  float* hB    = carve((size_t)N_ * DIM_ * 4);
  float* stats = carve(64 * 4);
  float* xd    = carve((size_t)B_ * MAXN_ * 128 * 4);
  float* traw  = carve((size_t)B_ * L_ * 128 * 4);
  float* tmat  = carve((size_t)B_ * L_ * 128 * 4);
  float* wbxT  = carve((size_t)B_ * SPAD_ * 128 * 4);
  float* Cb    = carve((size_t)B_ * L_ * SPAD_ * 4);
  float* wcx   = carve((size_t)B_ * 32 * MAXN_ * 4);
  float* wpt   = carve((size_t)B_ * 32 * L_ * 4);
  float* cvec  = carve((size_t)B_ * 128 * 4);
  float* pvec  = carve((size_t)B_ * 128 * 4);

  float* out0  = (float*)d_out;
  float* ac    = out0 + B_;
  float* ap    = ac + (size_t)B_ * MAXN_;

  // ---- GIN stack (5 layers) ----
  const float* hin = x;
  float* hout = hA;
  for (int layer = 0; layer < 5; ++layer) {
    int fin = (layer == 0) ? NFX_ : DIM_;
    const float* w1 = (const float*)d_in[5 + 6 * layer + 0];
    const float* b1 = (const float*)d_in[5 + 6 * layer + 1];
    const float* w2 = (const float*)d_in[5 + 6 * layer + 2];
    const float* b2 = (const float*)d_in[5 + 6 * layer + 3];
    const float* gm = (const float*)d_in[5 + 6 * layer + 4];
    const float* bt = (const float*)d_in[5 + 6 * layer + 5];
    int nel = N_ * fin;
    fill0_kernel<<<(nel + 255) / 256, 256, 0, stream>>>(agg, nel);
    gin_agg_kernel<<<E_, 128, 0, stream>>>(hin, ei, ei + E_, agg, fin);
    gin_mlp_kernel<<<N_ / 256, 256, 0, stream>>>(hin, agg, w1, b1, w2, b2, zbuf, fin);
    bn_stats_kernel<<<DIM_, 256, 0, stream>>>(zbuf, stats);
    bn_apply_kernel<<<(N_ * DIM_) / 256, 256, 0, stream>>>(zbuf, stats, gm, bt, hout);
    hin = hout;
    hout = (hout == hA) ? hB : hA;
  }

  // ---- compound dense branch ----
  xd_kernel<<<B_ * MAXN_, 128, 0, stream>>>(hin, fc1w, fc1b, xd);

  // ---- protein branch: big WMMA GEMM + LN ----
  prot_gemm_kernel<<<dim3(B_, L_ / 128), 256, 0, stream>>>(prot, tid, abw, traw);
  ln_kernel<<<(B_ * L_) / 8, 256, 0, stream>>>(traw, abb, lng, lnb, tmat);

  // ---- co-attention ----
  wbx_kernel<<<B_ * SPAD_, 128, 0, stream>>>(xd, Wb, wbxT);
  cgemm_kernel<<<dim3(B_, L_ / 128), 256, 0, stream>>>(tmat, wbxT, Cb);
  wcx_kernel<<<(B_ * 32 * MAXN_ + 255) / 256, 256, 0, stream>>>(xd, Wc, wcx);
  wpt_kernel<<<(B_ * 32 * L_ + 255) / 256, 256, 0, stream>>>(tmat, Wp, wpt);
  attn_c_kernel<<<B_, 256, 0, stream>>>(xd, wcx, wpt, Cb, whc, ac, cvec);
  attn_p_kernel<<<B_, 256, 0, stream>>>(tmat, wcx, wpt, Cb, whp, ap, pvec);

  // ---- head ----
  head_kernel<<<B_, 256, 0, stream>>>(cvec, pvec, faw, fab, fbw, fbb, ow, ob, out0);
}
